// CausalSelfAttention_17643725652393
// MI455X (gfx1250) — matrix operand order repaired and checked by reading, not compiled, not run on hardware
//
#include <hip/hip_runtime.h>

// ---------------------------------------------------------------------------
// CDNA5 (gfx1250) causal self-attention, bf16 WMMA pipeline.
//   K1: fp32 -> bf16 casts (x, w_attn, w_proj)
//   K2: QKV GEMM  — 64x64 block tile, weight tile async-staged to LDS
//       (global_load_async_to_lds_b128), double-buffered; writes
//       Q,K [B,H,T,hs], V^T [B,H,hs,T] (bf16)
//   K3: flash attention per 16-query tile (S wmma + online softmax + PV wmma)
//   K4: out projection GEMM (same LDS staging), fp32 output + bias
//
// WMMA register layouts per CDNA5 ISA 7.12.2 (wave32):
//   A 16x32 bf16 : lane m=lane&15, h=lane>>4; vgpr j<4 -> K=8h+2j, j>=4 -> K=16+8h+2(j-4)
//                  => two contiguous 16B runs per lane.
//   B 32x16 bf16 : lane n=lane&15, h=lane>>4; vgpr j -> same K as A vgpr j
//                  => two contiguous 16B runs per lane.
//   C/D 16x16 f32: lane n=lane&15; vgpr r -> row m = r + 8*(lane>>4), col n.
// ---------------------------------------------------------------------------

typedef __attribute__((ext_vector_type(16))) __bf16 v16bf;
typedef __attribute__((ext_vector_type(8)))  float  v8f;
typedef __attribute__((ext_vector_type(4)))  unsigned int u32x4;

static constexpr int Bb = 4, Tt = 2048, Cc = 768, Hh = 12, HS = 64;
static constexpr int M_ROWS = Bb * Tt;          // 8192
static constexpr int C3 = 3 * Cc;               // 2304
static constexpr int LDSP = 40;                 // padded LDS row (elements, 80B)

#if defined(__HIP_DEVICE_COMPILE__) && __has_builtin(__builtin_amdgcn_global_load_async_to_lds_b128)
#define HAVE_ASYNC_LDS 1
// builtin signature: (v4i addrspace(1)* src, v4i addrspace(3)* dst, Imm off, Imm cpol)
typedef int v4i __attribute__((vector_size(16)));
typedef __attribute__((address_space(1))) v4i* as1_v4i_p;
typedef __attribute__((address_space(3))) v4i* as3_v4i_p;
#else
#define HAVE_ASYNC_LDS 0
#endif

struct FragBF {
  union {
    u32x4 q[2];
    unsigned int u[8];
    v16bf v;
  };
};

__device__ __forceinline__ unsigned short f2bf(float x) {
  unsigned int u = __float_as_uint(x);
  u += 0x7FFFu + ((u >> 16) & 1u);              // round-to-nearest-even
  return (unsigned short)(u >> 16);
}

// A-matrix (16xK slice, row-major source, ld elements per row)
__device__ __forceinline__ FragBF load_a(const unsigned short* base, int ld,
                                         int row0, int c0, int lane) {
  int m = lane & 15, h = lane >> 4;
  const unsigned short* p = base + (size_t)(row0 + m) * ld + c0 + 8 * h;
  FragBF f;
  f.q[0] = *(const u32x4*)p;          // K = 8h+0..7   (rel c0)
  f.q[1] = *(const u32x4*)(p + 16);   // K = 16+8h+0..7
  return f;
}

// B-matrix (Kx16 slice; source row-major [N][K], ld elements per row)
__device__ __forceinline__ FragBF load_b(const unsigned short* base, int ld,
                                         int n0, int k0, int lane) {
  int n = lane & 15, h = lane >> 4;
  const unsigned short* p = base + (size_t)(n0 + n) * ld + k0 + 8 * h;
  FragBF f;
  f.q[0] = *(const u32x4*)p;          // K = 8h+0..7
  f.q[1] = *(const u32x4*)(p + 16);   // K = 16+8h+0..7
  return f;
}

// B-fragment read out of an LDS-staged 64x32 tile (rows padded to LDSP)
__device__ __forceinline__ FragBF load_b_lds(const unsigned short* tile,
                                             int n0f, int lane) {
  int n = lane & 15, h = lane >> 4;
  const unsigned short* p = tile + (n0f + n) * LDSP + 8 * h;
  FragBF f;
  f.q[0] = *(const u32x4*)p;
  f.q[1] = *(const u32x4*)(p + 16);
  return f;
}

__device__ __forceinline__ v8f wmma_bf16(const FragBF& a, const FragBF& b, v8f c) {
  return __builtin_amdgcn_wmma_f32_16x16x32_bf16(false, a.v, false, b.v,
                                                 (short)0, c, false, false);
}

// 16-byte global -> LDS copy (async on CDNA5 if builtin available)
__device__ __forceinline__ void copy16_g2l(const unsigned short* g, unsigned short* l) {
#if HAVE_ASYNC_LDS
  __builtin_amdgcn_global_load_async_to_lds_b128((as1_v4i_p)g, (as3_v4i_p)l, 0, 0);
#else
  u32x4 t = *(const u32x4*)g;
  *(u32x4*)l = t;
#endif
}

__device__ __forceinline__ void wait_async_copies() {
#if HAVE_ASYNC_LDS
#if __has_builtin(__builtin_amdgcn_s_wait_asynccnt)
  __builtin_amdgcn_s_wait_asynccnt(0);
#else
  asm volatile("s_wait_asynccnt 0x0" ::: "memory");
#endif
#endif
}

// stage one 64x32 bf16 weight tile (src row stride = Cc) into LDS buffer
// 256 16B chunks; 128 threads x 2 chunks
__device__ __forceinline__ void stage_btile(const unsigned short* wsrc,
                                            unsigned short* dst, int tid) {
  #pragma unroll
  for (int i = 0; i < 2; ++i) {
    int c = tid + i * 128;
    int row = c >> 2, part = c & 3;
    copy16_g2l(wsrc + (size_t)row * Cc + part * 8, dst + row * LDSP + part * 8);
  }
}

// ---------------------------------------------------------------------------
__global__ void cast_f32_bf16(const float* __restrict__ s,
                              unsigned short* __restrict__ d, int n) {
  int i = blockIdx.x * blockDim.x + threadIdx.x;
  if (i < n) d[i] = f2bf(s[i]);
}

// ---------------------------------------------------------------------------
// QKV GEMM: block = 64(M)x64(N) tile; 4 waves split M; shared B tile via LDS.
__global__ void qkv_gemm(const unsigned short* __restrict__ xb,
                         const unsigned short* __restrict__ wab,
                         const float* __restrict__ bias,
                         unsigned short* __restrict__ qbuf,
                         unsigned short* __restrict__ kbuf,
                         unsigned short* __restrict__ vtbuf) {
  __shared__ __align__(16) unsigned short btile[2][64 * LDSP];

  int tid  = threadIdx.x;
  int lane = tid & 31;
  int w    = tid >> 5;
  int mtb = blockIdx.x / 36, ns = blockIdx.x % 36;   // 128 x 36 blocks
  int row0 = mtb * 64 + w * 16, n0 = ns * 64;

  v8f acc[4];
  #pragma unroll
  for (int f = 0; f < 4; ++f) acc[f] = (v8f){0,0,0,0,0,0,0,0};

  stage_btile(wab + (size_t)n0 * Cc, btile[0], tid);

  for (int ks = 0; ks < Cc / 32; ++ks) {
    int kk = ks * 32;
    const unsigned short* cur = btile[ks & 1];
    wait_async_copies();
    __syncthreads();                               // tile[ks] visible to all
    if (ks + 1 < Cc / 32)                          // after barrier => WAR-safe
      stage_btile(wab + (size_t)n0 * Cc + (kk + 32), btile[(ks + 1) & 1], tid);

    FragBF a = load_a(xb, Cc, row0, kk, lane);
    #pragma unroll
    for (int f = 0; f < 4; ++f) {
      FragBF b = load_b_lds(cur, f * 16, lane);
      acc[f] = wmma_bf16(a, b, acc[f]);
    }
  }

  int n = lane & 15, h16 = lane >> 4;
  int sec = n0 / Cc;                       // entire 64-wide strip is in one of q/k/v
  #pragma unroll
  for (int f = 0; f < 4; ++f) {
    int o  = n0 + f * 16 + n;
    int oc = o % Cc;
    int hh = oc >> 6, d = oc & 63;
    float bv = bias[o];
    #pragma unroll
    for (int r = 0; r < 8; ++r) {
      int mrow = row0 + r + 8 * h16;
      int bidx = mrow >> 11, t = mrow & (Tt - 1);
      unsigned short val = f2bf(acc[f][r] + bv);
      size_t bh = (size_t)(bidx * Hh + hh);
      if (sec == 0)      qbuf[(bh * Tt + t) * HS + d] = val;
      else if (sec == 1) kbuf[(bh * Tt + t) * HS + d] = val;
      else               vtbuf[(bh * HS + d) * Tt + t] = val;   // V transposed
    }
  }
}

// ---------------------------------------------------------------------------
// Flash attention: one wave per (b,h, 16-query tile). Key blocks of 32.
// No block-wide barriers (waves have different causal trip counts).
__global__ void attn_flash(const unsigned short* __restrict__ qb,
                           const unsigned short* __restrict__ kb,
                           const unsigned short* __restrict__ vtb,
                           unsigned short* __restrict__ yb) {
  __shared__ unsigned short pbuf[4][16][32];   // per-wave P staging (C->A layout)

  int lane = threadIdx.x & 31;
  int w    = threadIdx.x >> 5;
  int wid  = blockIdx.x * 4 + w;
  int bh = wid >> 7;                 // 128 query tiles per (b,h)
  int qt = wid & 127;

  const unsigned short* qp = qb  + (size_t)bh * Tt * HS;
  const unsigned short* kp = kb  + (size_t)bh * Tt * HS;
  const unsigned short* vp = vtb + (size_t)bh * HS * Tt;

  int q0 = qt * 16;
  FragBF aq0 = load_a(qp, HS, q0, 0, lane);
  FragBF aq1 = load_a(qp, HS, q0, 32, lane);

  float mi[8], li[8];
  v8f O[4];
  #pragma unroll
  for (int r = 0; r < 8; ++r) { mi[r] = -1e30f; li[r] = 0.f; }
  #pragma unroll
  for (int f = 0; f < 4; ++f) O[f] = (v8f){0,0,0,0,0,0,0,0};

  int n = lane & 15, h16 = lane >> 4;
  const float scale = 0.125f;                  // 1/sqrt(64)
  int jend = qt / 2 + 1;                       // causal key-block extent

  for (int j = 0; j < jend; ++j) {
    int t0 = j * 32;
    if (j + 1 < jend) {                        // global_prefetch_b8 of next block
      __builtin_prefetch(kp + (size_t)(t0 + 32) * HS, 0, 0);
      __builtin_prefetch(vp + (t0 + 32), 0, 0);
    }

    // issue ALL global loads for this block first: K tiles then V tiles,
    // so V loads overlap the softmax math and the LDS round trip.
    FragBF bk0 = load_b(kp, HS, t0,      0,  lane);
    FragBF bk1 = load_b(kp, HS, t0 + 16, 0,  lane);
    FragBF bk2 = load_b(kp, HS, t0,      32, lane);
    FragBF bk3 = load_b(kp, HS, t0 + 16, 32, lane);
    FragBF bv0 = load_b(vp, Tt, 0,  t0, lane);
    FragBF bv1 = load_b(vp, Tt, 16, t0, lane);
    FragBF bv2 = load_b(vp, Tt, 32, t0, lane);
    FragBF bv3 = load_b(vp, Tt, 48, t0, lane);

    // S = Q(16x64) * K_block^T(64x32), f32
    v8f s[2];
    s[0] = (v8f){0,0,0,0,0,0,0,0};
    s[1] = (v8f){0,0,0,0,0,0,0,0};
    s[0] = wmma_bf16(aq0, bk0, s[0]);
    s[1] = wmma_bf16(aq0, bk1, s[1]);
    s[0] = wmma_bf16(aq1, bk2, s[0]);
    s[1] = wmma_bf16(aq1, bk3, s[1]);

    // scale + causal mask + row max (reduce across 16-lane half via shfl_xor)
    float pv0[8], pv1[8], mx[8];
    #pragma unroll
    for (int r = 0; r < 8; ++r) {
      int qrow = q0 + r + 8 * h16;
      float s0 = s[0][r] * scale; if (t0 + n      > qrow) s0 = -1e30f;
      float s1 = s[1][r] * scale; if (t0 + 16 + n > qrow) s1 = -1e30f;
      pv0[r] = s0; pv1[r] = s1;
      float m = fmaxf(s0, s1);
      m = fmaxf(m, __shfl_xor(m, 1));
      m = fmaxf(m, __shfl_xor(m, 2));
      m = fmaxf(m, __shfl_xor(m, 4));
      m = fmaxf(m, __shfl_xor(m, 8));
      mx[r] = m;
    }

    // online softmax update; stage P (bf16) into LDS in C-layout coordinates
    #pragma unroll
    for (int r = 0; r < 8; ++r) {
      float newm  = fmaxf(mi[r], mx[r]);
      float alpha = __expf(mi[r] - newm);
      mi[r] = newm;
      float p0 = __expf(pv0[r] - newm);
      float p1 = __expf(pv1[r] - newm);
      float sum = p0 + p1;
      sum += __shfl_xor(sum, 1);
      sum += __shfl_xor(sum, 2);
      sum += __shfl_xor(sum, 4);
      sum += __shfl_xor(sum, 8);
      li[r] = alpha * li[r] + sum;
      #pragma unroll
      for (int f = 0; f < 4; ++f) O[f][r] *= alpha;
      int mrow = r + 8 * h16;
      pbuf[w][mrow][n]      = f2bf(p0);
      pbuf[w][mrow][16 + n] = f2bf(p1);
    }

    // wave-private LDS round trip; DS ops are in-order per wave, wait stores
    asm volatile("s_wait_dscnt 0x0" ::: "memory");

    // reload P in A-fragment layout (16x32)
    FragBF pa;
    const unsigned int* prow = (const unsigned int*)&pbuf[w][n][0];
    #pragma unroll
    for (int jj = 0; jj < 8; ++jj) {
      int di = (jj < 4) ? (4 * h16 + jj) : (8 + 4 * h16 + (jj - 4));
      pa.u[jj] = prow[di];
    }

    // O += P(16x32) * V_block(32x64); V fragments already in registers
    O[0] = wmma_bf16(pa, bv0, O[0]);
    O[1] = wmma_bf16(pa, bv1, O[1]);
    O[2] = wmma_bf16(pa, bv2, O[2]);
    O[3] = wmma_bf16(pa, bv3, O[3]);
  }

  // normalize and write y as bf16 [B, T, C]
  int bidx = bh / Hh, hh = bh % Hh;
  #pragma unroll
  for (int r = 0; r < 8; ++r) {
    float inv = 1.0f / li[r];
    int t = q0 + r + 8 * h16;
    size_t rowbase = ((size_t)bidx * Tt + t) * Cc + hh * HS;
    #pragma unroll
    for (int f = 0; f < 4; ++f)
      yb[rowbase + f * 16 + n] = f2bf(O[f][r] * inv);
  }
}

// ---------------------------------------------------------------------------
// Output projection: 64x64 block tile, LDS-staged weights, fp32 out + bias.
__global__ void proj_gemm(const unsigned short* __restrict__ yb,
                          const unsigned short* __restrict__ wpb,
                          const float* __restrict__ bias,
                          float* __restrict__ out) {
  __shared__ __align__(16) unsigned short btile[2][64 * LDSP];

  int tid  = threadIdx.x;
  int lane = tid & 31;
  int w    = tid >> 5;
  int mtb = blockIdx.x / 12, ns = blockIdx.x % 12;   // 128 x 12 blocks
  int row0 = mtb * 64 + w * 16, n0 = ns * 64;

  v8f acc[4];
  #pragma unroll
  for (int f = 0; f < 4; ++f) acc[f] = (v8f){0,0,0,0,0,0,0,0};

  stage_btile(wpb + (size_t)n0 * Cc, btile[0], tid);

  for (int ks = 0; ks < Cc / 32; ++ks) {
    int kk = ks * 32;
    const unsigned short* cur = btile[ks & 1];
    wait_async_copies();
    __syncthreads();
    if (ks + 1 < Cc / 32)
      stage_btile(wpb + (size_t)n0 * Cc + (kk + 32), btile[(ks + 1) & 1], tid);

    FragBF a = load_a(yb, Cc, row0, kk, lane);
    #pragma unroll
    for (int f = 0; f < 4; ++f) {
      FragBF b = load_b_lds(cur, f * 16, lane);
      acc[f] = wmma_bf16(a, b, acc[f]);
    }
  }

  int n = lane & 15, h16 = lane >> 4;
  #pragma unroll
  for (int f = 0; f < 4; ++f) {
    int o = n0 + f * 16 + n;
    float bv = bias[o];
    #pragma unroll
    for (int r = 0; r < 8; ++r) {
      int mrow = row0 + r + 8 * h16;
      out[(size_t)mrow * Cc + o] = acc[f][r] + bv;
    }
  }
}

// ---------------------------------------------------------------------------
extern "C" void kernel_launch(void* const* d_in, const int* in_sizes, int n_in,
                              void* d_out, int out_size, void* d_ws, size_t ws_size,
                              hipStream_t stream) {
  const float* x      = (const float*)d_in[0];   // [4,2048,768]
  const float* w_attn = (const float*)d_in[1];   // [2304,768]
  const float* b_attn = (const float*)d_in[2];   // [2304]
  const float* w_proj = (const float*)d_in[3];   // [768,768]
  const float* b_proj = (const float*)d_in[4];   // [768]
  float* out = (float*)d_out;                    // [4,2048,768]

  // workspace layout (bytes)
  const size_t XB_E  = (size_t)M_ROWS * Cc;      // 6291456
  const size_t WAB_E = (size_t)C3 * Cc;          // 1769472
  const size_t WPB_E = (size_t)Cc * Cc;          //  589824
  const size_t QKV_E = (size_t)Bb * Hh * Tt * HS;// 6291456

  char* ws = (char*)d_ws;
  unsigned short* xb  = (unsigned short*)(ws);
  unsigned short* wab = (unsigned short*)(ws + 2 * XB_E);
  unsigned short* wpb = (unsigned short*)(ws + 2 * (XB_E + WAB_E));
  unsigned short* qb  = (unsigned short*)(ws + 2 * (XB_E + WAB_E + WPB_E));
  unsigned short* kb  = (unsigned short*)(ws + 2 * (XB_E + WAB_E + WPB_E + QKV_E));
  unsigned short* vtb = (unsigned short*)(ws + 2 * (XB_E + WAB_E + WPB_E + 2 * QKV_E));
  unsigned short* yb  = (unsigned short*)(ws + 2 * (XB_E + WAB_E + WPB_E + 3 * QKV_E));

  // K1: casts
  cast_f32_bf16<<<(int)((XB_E  + 255) / 256), 256, 0, stream>>>(x,      xb,  (int)XB_E);
  cast_f32_bf16<<<(int)((WAB_E + 255) / 256), 256, 0, stream>>>(w_attn, wab, (int)WAB_E);
  cast_f32_bf16<<<(int)((WPB_E + 255) / 256), 256, 0, stream>>>(w_proj, wpb, (int)WPB_E);

  // K2: QKV GEMM — 128 64-row tiles * 36 N-strips, 128 threads/block
  qkv_gemm<<<128 * 36, 128, 0, stream>>>(xb, wab, b_attn, qb, kb, vtb);

  // K3: flash attention — 4*12*128 = 6144 waves, 4 waves/block
  attn_flash<<<6144 / 4, 128, 0, stream>>>(qb, kb, vtb, yb);

  // K4: projection — 128 64-row tiles * 12 N-strips
  proj_gemm<<<128 * 12, 128, 0, stream>>>(yb, wpb, b_proj, out);
}
